// DecisionTransformerGPT2Attention_55327768708183
// MI455X (gfx1250) — matrix operand order repaired and checked by reading, not compiled, not run on hardware
//
#include <hip/hip_runtime.h>

#define BATCH   2
#define S_LEN   2048
#define D_MODEL 1024
#define NHEADS  16
#define HDIM    64
#define N3      3072   // 3*D

typedef __attribute__((ext_vector_type(16))) _Float16 v16h;
typedef __attribute__((ext_vector_type(8)))  _Float16 v8h;
typedef __attribute__((ext_vector_type(2)))  _Float16 v2h;
typedef __attribute__((ext_vector_type(8)))  float    v8f;
typedef unsigned int v4u __attribute__((ext_vector_type(4)));
typedef int          v4i __attribute__((ext_vector_type(4)));
typedef int          v8i __attribute__((ext_vector_type(8)));

#if defined(__has_builtin)
#if __has_builtin(__builtin_amdgcn_tensor_load_to_lds)
#define HAVE_TDM 1
#endif
#endif

// ---------------------------------------------------------------------------
// Tensor Data Mover: async 2D tile (16-bit elems) global -> LDS, with LDS row
// padding done by the TDM pad unit. Issue from one wave; wait TENSORcnt.
//   tile_w      : tile width in halfs (row length, multiple of 2)
//   rows        : tile height
//   mem_stride  : memory row stride in halfs
//   pad_int     : pad_interval code (2<<code dwords between pads)
//   pad_amt     : pad_amount code (code+1 dwords of pad)
// ---------------------------------------------------------------------------
#if defined(HAVE_TDM)
__device__ __forceinline__ void tdm_load_2d(unsigned int lds_addr,
                                            const void* gptr,
                                            unsigned int tile_w,
                                            unsigned int rows,
                                            unsigned int mem_stride,
                                            unsigned int pad_int,
                                            unsigned int pad_amt) {
  unsigned long long ga = (unsigned long long)(uintptr_t)gptr;
  v4u g0 = {1u,                               // count=1, user mode
            lds_addr,                         // lds_addr (bytes)
            (unsigned int)ga,                 // global_addr[31:0]
            (unsigned int)(ga >> 32) | (2u << 30)};  // addr[56:32] | type=2
  v8i g1 = {(int)((1u << 16)                  // data_size = 2 bytes
                  | (1u << 20)                // pad_enable
                  | (pad_int << 22) | (pad_amt << 25)),
            (int)(tile_w << 16),              // tensor_dim0 (lo16 in [31:16])
            (int)(rows << 16),                // tensor_dim1 (lo16 in [31:16])
            (int)(tile_w << 16),              // tile_dim0   in [31:16]
            (int)rows,                        // tile_dim1 ([15:0]), tile_dim2=0
            (int)mem_stride,                  // tensor_dim0_stride[31:0]
            0, 0};                            // stride hi / dim1_stride = 0
  v4i gz = {0, 0, 0, 0};
#if (__clang_major__ >= 23)
  v8i gz8 = {0, 0, 0, 0, 0, 0, 0, 0};
  __builtin_amdgcn_tensor_load_to_lds(g0, g1, gz, gz, gz8, 0);
#else
  __builtin_amdgcn_tensor_load_to_lds(g0, g1, gz, gz, 0);
#endif
}
#endif

// ---------------------------------------------------------------------------
// WMMA helpers (CDNA5 16x16x32 f16, f32 accumulate)
// ---------------------------------------------------------------------------
__device__ __forceinline__ v8f wmma_f16(v16h a, v16h b, v8f c) {
  return __builtin_amdgcn_wmma_f32_16x16x32_f16(false, a, false, b, (short)0, c,
                                                false, false);
}

// A fragment (16x32): lane L -> row mrow+(L&15); halfs 0..7 = K[kb+base..+7],
// halfs 8..15 = K[kb+base+16..+23], base = 0 (L<16) / 8 (L>=16).
__device__ __forceinline__ v16h load_a_frag(const _Float16* tile, int ld,
                                            int mrow, int kb) {
  const int lane = threadIdx.x & 31;
  const int m  = mrow + (lane & 15);
  const int c0 = kb + ((lane < 16) ? 0 : 8);
  const _Float16* p = tile + m * ld + c0;
  v8h lo = *(const v8h*)(p);
  v8h hi = *(const v8h*)(p + 16);
  v16h r;
#pragma unroll
  for (int i = 0; i < 8; ++i) { r[i] = lo[i]; r[i + 8] = hi[i]; }
  return r;
}

// B fragment (32x16) from an N-major tile: lane L -> col nb+(L&15);
// halfs 0..7 = K[kb+base..+7], halfs 8..15 = K[kb+base+16..+23], base = 0 (L<16) / 8 (L>=16).
__device__ __forceinline__ v16h load_b_frag(const _Float16* tileT, int ld,
                                            int nb, int kb) {
  const int lane = threadIdx.x & 31;
  const int n  = nb + (lane & 15);
  const int c0 = kb + ((lane < 16) ? 0 : 8);
  const _Float16* p = tileT + n * ld + c0;
  v8h lo = *(const v8h*)(p);
  v8h hi = *(const v8h*)(p + 16);
  v16h r;
#pragma unroll
  for (int i = 0; i < 8; ++i) { r[i] = lo[i]; r[i + 8] = hi[i]; }
  return r;
}

// ---------------------------------------------------------------------------
// Kernel 1: fused QKV projection. X[4096,1024]f32 @ W[1024,3072]f32 + b
//   -> Q,K,V in [B,H,S,hd] f16 (Q pre-scaled by 1/8).
// block tile 64(M) x 256(N), 256 threads = 8 waves (2x4), wave tile 32x64.
// ---------------------------------------------------------------------------
__global__ __launch_bounds__(256) void qkv_kernel(
    const float* __restrict__ hs, const float* __restrict__ w,
    const float* __restrict__ bias, _Float16* __restrict__ Qo,
    _Float16* __restrict__ Ko, _Float16* __restrict__ Vo) {
  __shared__ _Float16 As[64 * 40];     // [m][k]
  __shared__ _Float16 Bs[256 * 40];    // [n][k] (transposed)

  const int tid  = threadIdx.x;
  const int lane = tid & 31;
  const int wv   = tid >> 5;
  const int mw   = (wv >> 2) * 32;
  const int nw   = (wv & 3) * 64;
  const int m0   = blockIdx.y * 64;
  const int n0   = blockIdx.x * 256;

  const int arow = tid >> 2, acs = (tid & 3) * 8;        // A: 8 elems/thread
  const int bkr  = (tid >> 4) * 2, bnc = (tid & 15) * 16;  // B: 2x16/thread

  v8f acc[2][4] = {};

  for (int k0 = 0; k0 < D_MODEL; k0 += 32) {
    __syncthreads();
    {  // stage A (f32 -> f16), 16B vector LDS stores
      const float* gp = hs + (size_t)(m0 + arow) * D_MODEL + k0 + acs;
      v8h hv;
#pragma unroll
      for (int e = 0; e < 8; ++e) hv[e] = (_Float16)gp[e];
      *(v8h*)&As[arow * 40 + acs] = hv;
    }
    {  // stage B transposed: 2 k-rows packed -> ds_store_b32
      const float* g0p = w + (size_t)(k0 + bkr) * N3 + n0 + bnc;
      const float* g1p = g0p + N3;
#pragma unroll
      for (int e = 0; e < 16; ++e) {
        v2h pk = {(_Float16)g0p[e], (_Float16)g1p[e]};
        *(v2h*)&Bs[(bnc + e) * 40 + bkr] = pk;
      }
    }
    if (k0 + 32 < D_MODEL) {  // global_prefetch_b8 of next K-slab
      __builtin_prefetch(hs + (size_t)(m0 + arow) * D_MODEL + k0 + 32 + acs, 0, 0);
      __builtin_prefetch(w + (size_t)(k0 + 32 + bkr) * N3 + n0 + bnc, 0, 0);
    }
    __syncthreads();

    v16h a0 = load_a_frag(As, 40, mw, 0);
    v16h a1 = load_a_frag(As, 40, mw + 16, 0);
#pragma unroll
    for (int ni = 0; ni < 4; ++ni) {
      v16h b = load_b_frag(Bs, 40, nw + ni * 16, 0);
      acc[0][ni] = wmma_f16(a0, b, acc[0][ni]);
      acc[1][ni] = wmma_f16(a1, b, acc[1][ni]);
    }
  }

  // epilogue: bias + scatter into head-split Q/K/V (f16)
  const int g = (lane >> 4) & 1;
  const int region = n0 / D_MODEL;  // uniform per block (256 | 1024)
  _Float16* dst = (region == 0) ? Qo : ((region == 1) ? Ko : Vo);
  const float mul = (region == 0) ? 0.125f : 1.0f;  // 1/sqrt(64)

#pragma unroll
  for (int mi = 0; mi < 2; ++mi)
#pragma unroll
    for (int ni = 0; ni < 4; ++ni) {
      const int n  = n0 + nw + ni * 16 + (lane & 15);
      const float bv = bias[n];
      const int dD = n - region * D_MODEL;
      const int h = dD >> 6, dd = dD & 63;
#pragma unroll
      for (int r = 0; r < 8; ++r) {
        const int row = m0 + mw + mi * 16 + r + 8 * g;
        const int b = row >> 11, s = row & (S_LEN - 1);
        const float v = (acc[mi][ni][r] + bv) * mul;
        dst[(((size_t)(b * NHEADS + h)) * S_LEN + s) * HDIM + dd] = (_Float16)v;
      }
    }
}

// ---------------------------------------------------------------------------
// Kernel 2: flash attention. One block per (b,h,q-block of 64 rows).
// 128 threads = 4 waves, wave owns 16 q-rows. Q/K tiles via TDM async DMA.
// ---------------------------------------------------------------------------
__global__ __launch_bounds__(128) void attn_kernel(
    const _Float16* __restrict__ Q, const _Float16* __restrict__ K,
    const _Float16* __restrict__ V, _Float16* __restrict__ AO) {
  __shared__ _Float16 Qs[64 * 72];   // [qrow][d]
  __shared__ _Float16 Ks[64 * 72];   // [kcol][d]  (== B^T for Q.K^T)
  __shared__ _Float16 Vst[64 * 72];  // [d][k]     (== B^T for P.V)
  __shared__ _Float16 Ps[64 * 72];   // [qrow][k]  (A for P.V)

  const int tid  = threadIdx.x;
  const int lane = tid & 31;
  const int wv   = tid >> 5;
  const int g    = (lane >> 4) & 1;
  const int i    = blockIdx.x;   // q block index
  const int bh   = blockIdx.y;
  const int b    = bh >> 4, h = bh & 15;

  const _Float16* Qh = Q + (size_t)bh * S_LEN * HDIM;
  const _Float16* Kh = K + (size_t)bh * S_LEN * HDIM;
  const _Float16* Vh = V + (size_t)bh * S_LEN * HDIM;

  // ---- stage Q block (TDM: 64x64 halfs, pad 32dw-interval +4dw -> ld 72) ---
#if defined(HAVE_TDM)
  if (tid == 0) {
    tdm_load_2d((unsigned int)(uintptr_t)Qs, Qh + (size_t)i * 64 * HDIM,
                64, 64, 64, /*pad_int=*/4, /*pad_amt=*/3);
    __builtin_amdgcn_s_wait_tensorcnt(0);
  }
#else
  {
    const int row = tid >> 1, c0 = (tid & 1) * 32;
    const _Float16* gp = Qh + (size_t)(i * 64 + row) * HDIM + c0;
#pragma unroll
    for (int e = 0; e < 32; e += 8)
      *(v8h*)&Qs[row * 72 + c0 + e] = *(const v8h*)(gp + e);
  }
#endif
  __syncthreads();

  v16h aq0 = load_a_frag(Qs, 72, 16 * wv, 0);
  v16h aq1 = load_a_frag(Qs, 72, 16 * wv, 32);

  float mrun[8], lrun[8];
#pragma unroll
  for (int r = 0; r < 8; ++r) { mrun[r] = -3.0e38f; lrun[r] = 0.0f; }
  v8f of[4] = {};

  for (int j = 0; j <= i; ++j) {
    __syncthreads();
#if defined(HAVE_TDM)
    if (tid == 0)
      tdm_load_2d((unsigned int)(uintptr_t)Ks, Kh + (size_t)j * 64 * HDIM,
                  64, 64, 64, 4, 3);
#else
    {
      const int row = tid >> 1, c0 = (tid & 1) * 32;
      const _Float16* gk = Kh + (size_t)(j * 64 + row) * HDIM + c0;
#pragma unroll
      for (int e = 0; e < 32; e += 8)
        *(v8h*)&Ks[row * 72 + c0 + e] = *(const v8h*)(gk + e);
    }
#endif
    {  // stage V transposed: 2 k-rows packed -> ds_store_b32
      const int kr = (tid >> 2) * 2, c0 = (tid & 3) * 16;
      const _Float16* gv0 = Vh + (size_t)(j * 64 + kr) * HDIM + c0;
      const _Float16* gv1 = gv0 + HDIM;
#pragma unroll
      for (int e = 0; e < 16; ++e) {
        v2h pk = {gv0[e], gv1[e]};
        *(v2h*)&Vst[(c0 + e) * 72 + kr] = pk;
      }
    }
#if defined(HAVE_TDM)
    if (tid == 0) __builtin_amdgcn_s_wait_tensorcnt(0);
#endif
    __syncthreads();

    // scores S = Q . K^T  (16 q-rows x 64 k-cols per wave)
    v8f sc[4] = {};
#pragma unroll
    for (int nt = 0; nt < 4; ++nt) {
      v16h bk0 = load_b_frag(Ks, 72, nt * 16, 0);
      v16h bk1 = load_b_frag(Ks, 72, nt * 16, 32);
      sc[nt] = wmma_f16(aq0, bk0, sc[nt]);
      sc[nt] = wmma_f16(aq1, bk1, sc[nt]);
    }

    if (j == i) {  // causal mask only on diagonal block
#pragma unroll
      for (int nt = 0; nt < 4; ++nt) {
        const int col = j * 64 + nt * 16 + (lane & 15);
#pragma unroll
        for (int r = 0; r < 8; ++r) {
          const int rowq = i * 64 + 16 * wv + r + 8 * g;
          if (col > rowq) sc[nt][r] = -1.0e30f;
        }
      }
    }

    // online softmax (each row lives across one 16-lane half-group)
#pragma unroll
    for (int r = 0; r < 8; ++r) {
      float mx = fmaxf(fmaxf(sc[0][r], sc[1][r]), fmaxf(sc[2][r], sc[3][r]));
#pragma unroll
      for (int msk = 8; msk >= 1; msk >>= 1)
        mx = fmaxf(mx, __shfl_xor(mx, msk, 32));
      const float mnew  = fmaxf(mrun[r], mx);
      const float scale = __expf(mrun[r] - mnew);
      float rs = 0.0f;
      const int prow = (16 * wv + r + 8 * g) * 72;
#pragma unroll
      for (int nt = 0; nt < 4; ++nt) {
        const float p = __expf(sc[nt][r] - mnew);
        Ps[prow + nt * 16 + (lane & 15)] = (_Float16)p;
        rs += p;
      }
#pragma unroll
      for (int msk = 8; msk >= 1; msk >>= 1) rs += __shfl_xor(rs, msk, 32);
      lrun[r] = lrun[r] * scale + rs;
      mrun[r] = mnew;
#pragma unroll
      for (int nt = 0; nt < 4; ++nt) of[nt][r] *= scale;
    }
    __syncthreads();  // make cross-lane P writes visible to fragment loads

    // O += P . V
    v16h ap0 = load_a_frag(Ps, 72, 16 * wv, 0);
    v16h ap1 = load_a_frag(Ps, 72, 16 * wv, 32);
#pragma unroll
    for (int nt = 0; nt < 4; ++nt) {
      v16h bv0 = load_b_frag(Vst, 72, nt * 16, 0);
      v16h bv1 = load_b_frag(Vst, 72, nt * 16, 32);
      of[nt] = wmma_f16(ap0, bv0, of[nt]);
      of[nt] = wmma_f16(ap1, bv1, of[nt]);
    }
  }

  // normalize and write merged-heads output [B,S,D] f16
#pragma unroll
  for (int r = 0; r < 8; ++r) {
    const float inv  = 1.0f / lrun[r];
    const int   srow = i * 64 + 16 * wv + r + 8 * g;
#pragma unroll
    for (int nt = 0; nt < 4; ++nt) {
      const int dcol = h * HDIM + nt * 16 + (lane & 15);
      AO[((size_t)b * S_LEN + srow) * D_MODEL + dcol] =
          (_Float16)(of[nt][r] * inv);
    }
  }
}

// ---------------------------------------------------------------------------
// Kernel 3: output projection. AO[4096,1024]f16 @ Wp[1024,1024]f32 + b -> f32
// block tile 64x256; A tile staged by TDM (f16 source, pad 16dw +4dw -> ld 40)
// ---------------------------------------------------------------------------
__global__ __launch_bounds__(256) void proj_kernel(
    const _Float16* __restrict__ A, const float* __restrict__ w,
    const float* __restrict__ bias, float* __restrict__ out) {
  __shared__ _Float16 As[64 * 40];
  __shared__ _Float16 Bs[256 * 40];

  const int tid  = threadIdx.x;
  const int lane = tid & 31;
  const int wv   = tid >> 5;
  const int mw   = (wv >> 2) * 32;
  const int nw   = (wv & 3) * 64;
  const int m0   = blockIdx.y * 64;
  const int n0   = blockIdx.x * 256;

  const int arow = tid >> 2, acs = (tid & 3) * 8;
  const int bkr  = (tid >> 4) * 2, bnc = (tid & 15) * 16;

  v8f acc[2][4] = {};

  for (int k0 = 0; k0 < D_MODEL; k0 += 32) {
    __syncthreads();
#if defined(HAVE_TDM)
    if (tid == 0)
      tdm_load_2d((unsigned int)(uintptr_t)As,
                  A + (size_t)m0 * D_MODEL + k0, 32, 64, D_MODEL,
                  /*pad_int=*/3, /*pad_amt=*/3);
#else
    *(v8h*)&As[arow * 40 + acs] =
        *(const v8h*)(A + (size_t)(m0 + arow) * D_MODEL + k0 + acs);
#endif
    {  // stage B transposed: 2 k-rows packed -> ds_store_b32
      const float* g0p = w + (size_t)(k0 + bkr) * D_MODEL + n0 + bnc;
      const float* g1p = g0p + D_MODEL;
#pragma unroll
      for (int e = 0; e < 16; ++e) {
        v2h pk = {(_Float16)g0p[e], (_Float16)g1p[e]};
        *(v2h*)&Bs[(bnc + e) * 40 + bkr] = pk;
      }
    }
    if (k0 + 32 < D_MODEL)
      __builtin_prefetch(w + (size_t)(k0 + 32 + bkr) * D_MODEL + n0 + bnc, 0, 0);
#if defined(HAVE_TDM)
    if (tid == 0) __builtin_amdgcn_s_wait_tensorcnt(0);
#endif
    __syncthreads();

    v16h a0 = load_a_frag(As, 40, mw, 0);
    v16h a1 = load_a_frag(As, 40, mw + 16, 0);
#pragma unroll
    for (int ni = 0; ni < 4; ++ni) {
      v16h bfr = load_b_frag(Bs, 40, nw + ni * 16, 0);
      acc[0][ni] = wmma_f16(a0, bfr, acc[0][ni]);
      acc[1][ni] = wmma_f16(a1, bfr, acc[1][ni]);
    }
  }

  const int g = (lane >> 4) & 1;
#pragma unroll
  for (int mi = 0; mi < 2; ++mi)
#pragma unroll
    for (int ni = 0; ni < 4; ++ni) {
      const int n  = n0 + nw + ni * 16 + (lane & 15);
      const float bv = bias[n];
#pragma unroll
      for (int r = 0; r < 8; ++r) {
        const int row = m0 + mw + mi * 16 + r + 8 * g;
        out[(size_t)row * D_MODEL + n] = acc[mi][ni][r] + bv;
      }
    }
}

// ---------------------------------------------------------------------------
extern "C" void kernel_launch(void* const* d_in, const int* in_sizes, int n_in,
                              void* d_out, int out_size, void* d_ws,
                              size_t ws_size, hipStream_t stream) {
  const float* hs = (const float*)d_in[0];  // [B,S,D]
  const float* aw = (const float*)d_in[1];  // [D,3D]
  const float* ab = (const float*)d_in[2];  // [3D]
  const float* pw = (const float*)d_in[3];  // [D,D]
  const float* pb = (const float*)d_in[4];  // [D]
  float* out = (float*)d_out;               // [B,S,D]

  const size_t per = (size_t)BATCH * NHEADS * S_LEN * HDIM;  // 4M halfs
  _Float16* Q  = (_Float16*)d_ws;
  _Float16* Kb = Q + per;
  _Float16* Vb = Kb + per;
  _Float16* AO = Vb + per;  // [B,S,D] f16; total ws use = 32 MB

  qkv_kernel<<<dim3(N3 / 256, (BATCH * S_LEN) / 64), 256, 0, stream>>>(
      hs, aw, ab, Q, Kb, Vb);
  attn_kernel<<<dim3(S_LEN / 64, BATCH * NHEADS), 128, 0, stream>>>(
      Q, Kb, Vb, AO);
  proj_kernel<<<dim3(D_MODEL / 256, (BATCH * S_LEN) / 64), 256, 0, stream>>>(
      AO, pw, pb, out);
}
